// UTransformer_12292196401464
// MI455X (gfx1250) — compile-verified, matrix operand order checked by reading, not run on hardware
//
#include <hip/hip_runtime.h>
#include <math.h>

typedef __attribute__((ext_vector_type(16))) __bf16 bf16x16;
typedef __attribute__((ext_vector_type(2)))  __bf16 bf16x2;
typedef __attribute__((ext_vector_type(2)))  float  f32x2;
typedef __attribute__((ext_vector_type(8)))  float  v8f;

#define TILE_M 128
#define TILE_N 64
#define TILE_K 32
#define LDST   40   // padded LDS row stride in ushorts (80 B: 16B-aligned, conflict-free)

// ---------------------------------------------------------------------------
// f32 -> bf16 RNE converts; pair form maps directly to v_cvt_pk_bf16_f32
// ---------------------------------------------------------------------------
__device__ __forceinline__ unsigned short f2bf(float f) {
  union { __bf16 h; unsigned short s; } c;
  c.h = (__bf16)f;
  return c.s;
}
__device__ __forceinline__ unsigned int f2bf2(float lo, float hi) {
  f32x2 f; f.x = lo; f.y = hi;
  union { bf16x2 h; unsigned int u; } c;
  c.h = __builtin_convertvector(f, bf16x2);
  return c.u;
}

// ---------------------------------------------------------------------------
// General batched GEMM: C = alpha * (A @ B) [+bias] [+residual] [relu]
//   A: f32, row-major MxK (lda), or conv-gather mode (K = 3*Cin, per-batch pad)
//   B: f32, row-major KxN (ldb) or transposed NxK (ldb) when bTrans
//   Math: bf16 WMMA 16x16x32, f32 accumulate; double-buffered LDS pipeline
//   Batch z: offset = (z >> zShift)*sXo + (z & (2^zShift - 1))*sXi
// ---------------------------------------------------------------------------
struct GemmParams {
  const float* A; const float* B; float* C;
  const float* bias; const float* residual;
  int M, N, K;
  int lda, ldb, ldc, ldr;
  long long sAo, sAi, sBo, sBi, sCo, sCi;
  int zShift;    // log2(inner batch count); 0 for non-batched
  float alpha;
  int bTrans;    // B stored N x K row-major
  int convMode;  // A gathered as kernel-3 conv over Cin channels, L=512/batch
  int cinShift;  // log2(Cin); Cin is a power of two (1024 or 4096)
  int relu;
};

__global__ void __launch_bounds__(128) gemm_bf16_wmma(GemmParams p) {
  __shared__ __align__(16) unsigned short As[2][TILE_M * LDST];
  __shared__ __align__(16) unsigned short Bs[2][TILE_N * LDST];

  const int bz = blockIdx.z;
  const int zo = bz >> p.zShift;
  const int zi = bz & ((1 << p.zShift) - 1);
  const float* A = p.A + zo * p.sAo + zi * p.sAi;
  const float* B = p.B + zo * p.sBo + zi * p.sBi;
  float*       C = p.C + zo * p.sCo + zi * p.sCi;

  const int m0 = blockIdx.y * TILE_M;
  const int n0 = blockIdx.x * TILE_N;
  const int tid  = threadIdx.x;
  const int wave = tid >> 5;
  const int lane = tid & 31;
  const int wm = wave >> 1;   // 0..1 -> 64-row strip
  const int wn = wave & 1;    // 0..1 -> 32-col strip
  const int cinMask = (1 << p.cinShift) - 1;

  // ---- staging helpers (8 float4 for A, 4 float4 for B per thread) --------
  auto loadA = [&](int k0, float4* reg) {
    #pragma unroll
    for (int it = 0; it < 8; ++it) {
      int idx = tid + (it << 7);          // 0..1023
      int r   = idx >> 3;                 // 0..127
      int c4  = (idx & 7) << 2;           // 0,4,..,28
      int gm  = m0 + r;
      float4 val = make_float4(0.f, 0.f, 0.f, 0.f);
      if (!p.convMode) {
        val = *(const float4*)(A + (long long)gm * p.lda + (k0 + c4));
      } else {
        int b   = gm >> 9;                // token -> batch
        int t   = gm & 511;               // token -> position
        int kk  = k0 + c4;
        int tap = kk >> p.cinShift;       // 0..2 (tile never crosses tap edge)
        int ch  = kk & cinMask;
        int tt  = t + tap - 1;
        if (tt >= 0 && tt < 512)
          val = *(const float4*)(A + (((long long)((b << 9) + tt)) << p.cinShift) + ch);
      }
      reg[it] = val;
    }
  };
  auto storeA = [&](int buf, const float4* reg) {
    #pragma unroll
    for (int it = 0; it < 8; ++it) {
      int idx = tid + (it << 7);
      int r   = idx >> 3;
      int c4  = (idx & 7) << 2;
      uint2 pk;
      pk.x = f2bf2(reg[it].x, reg[it].y);
      pk.y = f2bf2(reg[it].z, reg[it].w);
      *(uint2*)&As[buf][r * LDST + c4] = pk;   // 8B-aligned (80B row stride)
    }
  };
  auto loadB = [&](int k0, float4* reg) {
    if (!p.bTrans) {
      #pragma unroll
      for (int it = 0; it < 4; ++it) {
        int idx = tid + (it << 7);        // 0..511
        int kk  = idx >> 4;               // 0..31
        int n4  = (idx & 15) << 2;        // 0,4,..,60
        reg[it] = *(const float4*)(B + (long long)(k0 + kk) * p.ldb + (n0 + n4));
      }
    } else {
      #pragma unroll
      for (int it = 0; it < 4; ++it) {
        int idx = tid + (it << 7);
        int nn  = idx >> 3;               // 0..63
        int k4  = (idx & 7) << 2;         // 0,4,..,28
        reg[it] = *(const float4*)(B + (long long)(n0 + nn) * p.ldb + (k0 + k4));
      }
    }
  };
  auto storeB = [&](int buf, const float4* reg) {
    if (!p.bTrans) {
      #pragma unroll
      for (int it = 0; it < 4; ++it) {
        int idx = tid + (it << 7);
        int kk  = idx >> 4;
        int n4  = (idx & 15) << 2;
        Bs[buf][(n4 + 0) * LDST + kk] = f2bf(reg[it].x);
        Bs[buf][(n4 + 1) * LDST + kk] = f2bf(reg[it].y);
        Bs[buf][(n4 + 2) * LDST + kk] = f2bf(reg[it].z);
        Bs[buf][(n4 + 3) * LDST + kk] = f2bf(reg[it].w);
      }
    } else {
      #pragma unroll
      for (int it = 0; it < 4; ++it) {
        int idx = tid + (it << 7);
        int nn  = idx >> 3;
        int k4  = (idx & 7) << 2;
        uint2 pk;
        pk.x = f2bf2(reg[it].x, reg[it].y);
        pk.y = f2bf2(reg[it].z, reg[it].w);
        *(uint2*)&Bs[buf][nn * LDST + k4] = pk;
      }
    }
  };

  v8f acc[4][2];
  const v8f vzero = {0.f,0.f,0.f,0.f,0.f,0.f,0.f,0.f};
  #pragma unroll
  for (int mt = 0; mt < 4; ++mt) { acc[mt][0] = vzero; acc[mt][1] = vzero; }

  union Frag { bf16x16 v; uint4 q[2]; };

  float4 aReg[8], bReg[4];
  loadA(0, aReg); loadB(0, bReg);
  storeA(0, aReg); storeB(0, bReg);
  __syncthreads();

  const int lr = lane & 15;
  const int kb = (lane >> 4) << 3;        // 0 or 8 (ISA 16-bit operand layout)

  int cur = 0;
  for (int k0 = 0; k0 < p.K; k0 += TILE_K) {
    const bool more = (k0 + TILE_K) < p.K;
    if (more) { loadA(k0 + TILE_K, aReg); loadB(k0 + TILE_K, bReg); }

    // fragments: lanes0-15 K{kb..kb+7, kb+16..kb+23}; lanes16-31 kb=8
    Frag af[4], bfr[2];
    #pragma unroll
    for (int mt = 0; mt < 4; ++mt) {
      int row = (wm << 6) + (mt << 4) + lr;
      const unsigned short* sp = &As[cur][row * LDST + kb];
      af[mt].q[0] = *(const uint4*)(sp);
      af[mt].q[1] = *(const uint4*)(sp + 16);
    }
    #pragma unroll
    for (int nt = 0; nt < 2; ++nt) {
      int col = (wn << 5) + (nt << 4) + lr;
      const unsigned short* sp = &Bs[cur][col * LDST + kb];
      bfr[nt].q[0] = *(const uint4*)(sp);
      bfr[nt].q[1] = *(const uint4*)(sp + 16);
    }
    #pragma unroll
    for (int mt = 0; mt < 4; ++mt)
      #pragma unroll
      for (int nt = 0; nt < 2; ++nt)
        acc[mt][nt] = __builtin_amdgcn_wmma_f32_16x16x32_bf16(
            false, af[mt].v, false, bfr[nt].v, (short)0, acc[mt][nt],
            false, false);

    if (more) { storeA(cur ^ 1, aReg); storeB(cur ^ 1, bReg); }
    __syncthreads();
    cur ^= 1;
  }

  // ---- epilogue: D layout: VGPR r -> row r (+8 for lanes 16-31), col lane&15
  const int ra = (lane >> 4) << 3;
  #pragma unroll
  for (int mt = 0; mt < 4; ++mt) {
    #pragma unroll
    for (int nt = 0; nt < 2; ++nt) {
      int colg = n0 + (wn << 5) + (nt << 4) + lr;
      #pragma unroll
      for (int r = 0; r < 8; ++r) {
        int rowg = m0 + (wm << 6) + (mt << 4) + r + ra;
        float val = acc[mt][nt][r] * p.alpha;
        if (p.bias)     val += p.bias[colg];
        if (p.residual) val += p.residual[(long long)rowg * p.ldr + colg];
        if (p.relu)     val  = fmaxf(val, 0.f);
        C[(long long)rowg * p.ldc + colg] = val;
      }
    }
  }
}

// ---------------------------------------------------------------------------
// Elementwise / reduction kernels
// ---------------------------------------------------------------------------
#define NTOK 4096
#define DMODEL 1024

__global__ void embed_init(const int* __restrict__ X, const float* __restrict__ emb,
                           float* state, float* prev, float* hp, float* rem, float* nup) {
  int tok = blockIdx.x;
  int id  = X[tok];
  for (int c = threadIdx.x; c < DMODEL; c += blockDim.x) {
    float v = emb[(long long)id * DMODEL + c];
    state[(long long)tok * DMODEL + c] = v;
    prev [(long long)tok * DMODEL + c] = 0.f;
  }
  if (threadIdx.x == 0) { hp[tok] = 0.f; rem[tok] = 0.f; nup[tok] = 0.f; }
}

__global__ void add_time_pos(float* state, int layer) {
  long long i = (long long)blockIdx.x * 256 + threadIdx.x; // over 4096*1024
  int c   = (int)(i & 1023);
  int tok = (int)(i >> 10);
  int t   = tok & 511;
  int ts  = (c < 512) ? c : c - 512;
  const float log_inc = 0.01802414935f;  // ln(10000)/511
  float inv = expf(-(float)ts * log_inc);
  float at = (float)t * inv, ap = (float)layer * inv;
  float vt = (c < 512) ? sinf(at) : cosf(at);
  float vp = (c < 512) ? sinf(ap) : cosf(ap);
  state[i] += vt + vp;
}

__global__ void act_halt(const float* __restrict__ state, const float* __restrict__ p_w,
                         const float* __restrict__ p_b,
                         float* hp, float* rem, float* nup, float* uw) {
  __shared__ float red[256];
  int tok = blockIdx.x, tid = threadIdx.x;
  const float* x = state + (long long)tok * DMODEL;
  float s = 0.f;
  for (int c = tid; c < DMODEL; c += 256) s += x[c] * p_w[c];
  red[tid] = s; __syncthreads();
  for (int st = 128; st > 0; st >>= 1) {
    if (tid < st) red[tid] += red[tid + st];
    __syncthreads();
  }
  if (tid == 0) {
    float pr = 1.f / (1.f + expf(-(red[0] + p_b[0])));
    float h = hp[tok], r = rem[tok], n = nup[tok];
    float still = (h < 1.0f) ? 1.f : 0.f;
    float add = h + pr * still;
    float nh  = (add > 0.9f)  ? still : 0.f;
    float st2 = (add <= 0.9f) ? still : 0.f;
    h = h + pr * st2;
    r = r + nh * (1.f - h);
    h = h + nh * r;
    n = n + st2 + nh;
    hp[tok] = h; rem[tok] = r; nup[tok] = n;
    uw[tok] = pr * st2 + nh * r;
  }
}

__global__ void layernorm_k(const float* __restrict__ x, const float* __restrict__ g,
                            const float* __restrict__ b, float* __restrict__ y) {
  __shared__ float rs[256], rq[256];
  int tok = blockIdx.x, tid = threadIdx.x;
  const float* xp = x + (long long)tok * DMODEL;
  float s = 0.f, q = 0.f;
  for (int c = tid; c < DMODEL; c += 256) { float v = xp[c]; s += v; q += v * v; }
  rs[tid] = s; rq[tid] = q; __syncthreads();
  for (int st = 128; st > 0; st >>= 1) {
    if (tid < st) { rs[tid] += rs[tid + st]; rq[tid] += rq[tid + st]; }
    __syncthreads();
  }
  float mu  = rs[0] * (1.f / 1024.f);
  float var = fmaxf((rq[0] - 1024.f * mu * mu) * (1.f / 1023.f), 0.f); // ddof=1
  float inv = 1.f / (sqrtf(var) + 1e-6f);
  float* yp = y + (long long)tok * DMODEL;
  for (int c = tid; c < DMODEL; c += 256) yp[c] = g[c] * (xp[c] - mu) * inv + b[c];
}

__global__ void softmax_rows(float* __restrict__ logits) {  // rows of 512
  __shared__ float red[256];
  long long row = blockIdx.x;
  float* x = logits + row * 512;
  int tid = threadIdx.x;
  float m = -INFINITY;
  for (int c = tid; c < 512; c += 256) m = fmaxf(m, x[c]);
  red[tid] = m; __syncthreads();
  for (int st = 128; st > 0; st >>= 1) {
    if (tid < st) red[tid] = fmaxf(red[tid], red[tid + st]);
    __syncthreads();
  }
  m = red[0]; __syncthreads();
  float s = 0.f;
  for (int c = tid; c < 512; c += 256) { float e = expf(x[c] - m); x[c] = e; s += e; }
  red[tid] = s; __syncthreads();
  for (int st = 128; st > 0; st >>= 1) {
    if (tid < st) red[tid] += red[tid + st];
    __syncthreads();
  }
  float invs = 1.f / red[0];
  for (int c = tid; c < 512; c += 256) x[c] *= invs;
}

__global__ void prev_update(const float* __restrict__ state, float* __restrict__ prev,
                            const float* __restrict__ uw) {
  long long i = (long long)blockIdx.x * 256 + threadIdx.x;
  int tok = (int)(i >> 10);
  float u = uw[tok];
  prev[i] = state[i] * u + prev[i] * (1.f - u);
}

__global__ void final_head(const float* __restrict__ prev, const float* __restrict__ W_out,
                           const float* __restrict__ b_out, float* __restrict__ out) {
  __shared__ float red[256];
  __shared__ float sh[4];
  int b = blockIdx.x, tid = threadIdx.x;
  float acc[4] = {0.f, 0.f, 0.f, 0.f};
  for (int c = tid; c < DMODEL; c += 256) {
    float s = 0.f;
    const float* base = prev + ((long long)b * 512) * DMODEL + c;
    for (int t = 0; t < 512; ++t) s += base[(long long)t * DMODEL];
    s *= (1.f / 512.f);
    #pragma unroll
    for (int j = 0; j < 4; ++j) acc[j] += s * W_out[c * 4 + j];
  }
  for (int j = 0; j < 4; ++j) {
    red[tid] = acc[j]; __syncthreads();
    for (int st = 128; st > 0; st >>= 1) {
      if (tid < st) red[tid] += red[tid + st];
      __syncthreads();
    }
    if (tid == 0) sh[j] = red[0] + b_out[j];
    __syncthreads();
  }
  if (tid == 0) {
    float m = fmaxf(fmaxf(sh[0], sh[1]), fmaxf(sh[2], sh[3]));
    float e[4], s = 0.f;
    #pragma unroll
    for (int j = 0; j < 4; ++j) { e[j] = expf(sh[j] - m); s += e[j]; }
    #pragma unroll
    for (int j = 0; j < 4; ++j) {
      out[b * 4 + j]      = sh[j];       // a_hat
      out[32 + b * 4 + j] = e[j] / s;    // softmax(a_hat)
    }
  }
}

__global__ void write_tail(const float* __restrict__ rem, const float* __restrict__ nup,
                           float* __restrict__ out) {
  int i = blockIdx.x * 256 + threadIdx.x;
  if (i < NTOK) {
    out[64 + i]        = rem[i];
    out[64 + NTOK + i] = nup[i];
  }
}

// ---------------------------------------------------------------------------
// Host orchestration
// ---------------------------------------------------------------------------
static inline void run_gemm(hipStream_t s, const GemmParams& p, int batches) {
  dim3 grid(p.N / TILE_N, p.M / TILE_M, batches);
  gemm_bf16_wmma<<<grid, dim3(128), 0, s>>>(p);
}

static inline GemmParams base_params() {
  GemmParams p = {};
  p.alpha = 1.f;
  return p;
}

extern "C" void kernel_launch(void* const* d_in, const int* in_sizes, int n_in,
                              void* d_out, int out_size, void* d_ws, size_t ws_size,
                              hipStream_t stream) {
  (void)in_sizes; (void)n_in; (void)out_size; (void)ws_size;
  const int*   X     = (const int*)  d_in[0];
  const float* emb   = (const float*)d_in[1];
  const float* p_w   = (const float*)d_in[2];
  const float* p_b   = (const float*)d_in[3];
  const float* Wq    = (const float*)d_in[4];
  const float* Wk    = (const float*)d_in[5];
  const float* Wv    = (const float*)d_in[6];
  const float* Wo    = (const float*)d_in[7];
  const float* ln1_g = (const float*)d_in[8];
  const float* ln1_b = (const float*)d_in[9];
  const float* ln2_g = (const float*)d_in[10];
  const float* ln2_b = (const float*)d_in[11];
  const float* K1    = (const float*)d_in[12];
  const float* c1_b  = (const float*)d_in[13];
  const float* K2    = (const float*)d_in[14];
  const float* c2_b  = (const float*)d_in[15];
  const float* W_out = (const float*)d_in[16];
  const float* b_out = (const float*)d_in[17];
  float* out = (float*)d_out;

  float* ws = (float*)d_ws;
  const long long ACT = (long long)NTOK * DMODEL;       // 4M floats
  float* state = ws;
  float* prev  = ws + 1 * ACT;
  float* xn    = ws + 2 * ACT;
  float* q     = ws + 3 * ACT;
  float* k     = ws + 4 * ACT;
  float* v     = ws + 5 * ACT;
  float* ctx   = ws + 6 * ACT;
  float* big   = ws + 7 * ACT;                          // logits (134MB) | ffn hidden (67MB)
  float* hp    = big + 33554432LL;                      // 128*512*512
  float* rem   = hp + NTOK;
  float* nup   = rem + NTOK;
  float* uw    = nup + NTOK;

  embed_init<<<NTOK, 256, 0, stream>>>(X, emb, state, prev, hp, rem, nup);

  for (int layer = 0; layer < 6; ++layer) {
    add_time_pos<<<16384, 256, 0, stream>>>(state, layer);
    act_halt<<<NTOK, 256, 0, stream>>>(state, p_w, p_b, hp, rem, nup, uw);
    layernorm_k<<<NTOK, 256, 0, stream>>>(state, ln1_g, ln1_b, xn);

    // --- Q/K/V projections (M=4096, K=1024, N=1024); 1/sqrt(64) folded into Q
    {
      GemmParams p = base_params();
      p.A = xn; p.lda = 1024; p.M = 4096; p.K = 1024; p.N = 1024;
      p.ldb = 1024; p.ldc = 1024;
      p.B = Wq; p.C = q; p.alpha = 0.125f;  run_gemm(stream, p, 1);
      p.B = Wk; p.C = k; p.alpha = 1.f;     run_gemm(stream, p, 1);
      p.B = Wv; p.C = v;                    run_gemm(stream, p, 1);
    }

    // --- logits[b,h] = q_bh (512x64) @ k_bh^T (64x512), 128 batches
    {
      GemmParams p = base_params();
      p.M = 512; p.N = 512; p.K = 64;
      p.A = q;   p.lda = 1024; p.sAo = 512LL * 1024; p.sAi = 64;
      p.B = k;   p.ldb = 1024; p.sBo = 512LL * 1024; p.sBi = 64; p.bTrans = 1;
      p.C = big; p.ldc = 512;  p.sCo = 16LL * 512 * 512; p.sCi = 512LL * 512;
      p.zShift = 4;
      run_gemm(stream, p, 128);
    }
    softmax_rows<<<65536, 256, 0, stream>>>(big);

    // --- ctx[b,h] = probs (512x512) @ v_bh (512x64)
    {
      GemmParams p = base_params();
      p.M = 512; p.N = 64; p.K = 512;
      p.A = big; p.lda = 512;  p.sAo = 16LL * 512 * 512; p.sAi = 512LL * 512;
      p.B = v;   p.ldb = 1024; p.sBo = 512LL * 1024;     p.sBi = 64;
      p.C = ctx; p.ldc = 1024; p.sCo = 512LL * 1024;     p.sCi = 64;
      p.zShift = 4;
      run_gemm(stream, p, 128);
    }

    // --- state = state + ctx @ Wo
    {
      GemmParams p = base_params();
      p.M = 4096; p.N = 1024; p.K = 1024;
      p.A = ctx; p.lda = 1024; p.B = Wo; p.ldb = 1024;
      p.C = state; p.ldc = 1024; p.residual = state; p.ldr = 1024;
      run_gemm(stream, p, 1);
    }

    layernorm_k<<<NTOK, 256, 0, stream>>>(state, ln2_g, ln2_b, xn);

    // --- h = relu(conv1d(xn, K1) + c1_b): conv-as-GEMM, K = 3*1024
    {
      GemmParams p = base_params();
      p.M = 4096; p.N = 4096; p.K = 3072;
      p.A = xn; p.convMode = 1; p.cinShift = 10; p.lda = 1024;
      p.B = K1; p.ldb = 4096;
      p.C = big; p.ldc = 4096; p.bias = c1_b; p.relu = 1;
      run_gemm(stream, p, 1);
    }
    // --- state = state + conv1d(h, K2) + c2_b: K = 3*4096
    {
      GemmParams p = base_params();
      p.M = 4096; p.N = 1024; p.K = 12288;
      p.A = big; p.convMode = 1; p.cinShift = 12; p.lda = 4096;
      p.B = K2; p.ldb = 1024;
      p.C = state; p.ldc = 1024; p.bias = c2_b;
      p.residual = state; p.ldr = 1024;
      run_gemm(stream, p, 1);
    }

    prev_update<<<16384, 256, 0, stream>>>(state, prev, uw);
  }

  final_head<<<8, 256, 0, stream>>>(prev, W_out, b_out, out);
  write_tail<<<16, 256, 0, stream>>>(rem, nup, out);
}
